// TransNovo_17179869184657
// MI455X (gfx1250) — compile-verified, matrix operand order checked by reading, not run on hardware
//
#include <hip/hip_runtime.h>
#include <hip/hip_bf16.h>

// Toggle: 1 = use CDNA5 async global->LDS copies in the GEMM (inline asm),
//         0 = classic global_load_b128 + ds_store_b128 staging.
#define GEMM_ASYNC_LDS 1

// ---------------------------------------------------------------------------
// Types for CDNA5 WMMA (wave32): D(16x16,f32) = A(16x32,f16) x B(32x16,f16) + C
// ---------------------------------------------------------------------------
typedef __attribute__((ext_vector_type(16))) _Float16 hvec16;
typedef __attribute__((ext_vector_type(8)))  _Float16 hvec8;
typedef __attribute__((ext_vector_type(8)))  float    fvec8;

__device__ __forceinline__ hvec16 frag_ld(const _Float16* p0, const _Float16* p1) {
    hvec8 lo = *(const hvec8*)p0;   // K = kb .. kb+7
    hvec8 hi = *(const hvec8*)p1;   // K = 16+kb .. 16+kb+7
    hvec16 r;
#pragma unroll
    for (int i = 0; i < 8; ++i) { r[i] = lo[i]; r[i + 8] = hi[i]; }
    return r;
}

__device__ __forceinline__ fvec8 wmma_f16(hvec16 a, hvec16 b, fvec8 c) {
    return __builtin_amdgcn_wmma_f32_16x16x32_f16(false, a, false, b, (short)0, c,
                                                  false, false);
}

// 16-byte global -> LDS copy. Async op on CDNA5 (ASYNCcnt-tracked).
__device__ __forceinline__ void copy16_g2l(const _Float16* g, _Float16* lds) {
#if GEMM_ASYNC_LDS
    unsigned l = (unsigned)(uintptr_t)lds;   // generic LDS ptr: low 32 bits = LDS offset
    asm volatile("global_load_async_to_lds_b128 %0, %1, off"
                 :: "v"(l), "v"(g) : "memory");
#else
    *(hvec8*)lds = *(const hvec8*)g;
#endif
}

__device__ __forceinline__ void copy16_wait() {
#if GEMM_ASYNC_LDS
    asm volatile("s_wait_asynccnt 0x0" ::: "memory");
#endif
}

// ---------------------------------------------------------------------------
// Small utility kernels
// ---------------------------------------------------------------------------
// (H, D, Dk) f32  ->  (H*Dk, D) f16, scaled   [pre-transposed GEMM B-operand]
__global__ void repack_qkv_kernel(const float* __restrict__ w, _Float16* __restrict__ o,
                                  int H, int D, int Dk, float scale) {
    int idx = blockIdx.x * 256 + threadIdx.x;     // over (H*Dk)*D, output [n][d]
    int tot = H * Dk * D;
    if (idx >= tot) return;
    int n = idx / D;
    int d = idx - n * D;
    int h = n / Dk, k = n - h * Dk;
    o[idx] = (_Float16)(w[((size_t)h * D + d) * Dk + k] * scale);
}

// (K, N) f32 -> (N, K) f16  transpose + convert  [weight -> GEMM B-operand]
__global__ void cvt_transpose_kernel(const float* __restrict__ a, _Float16* __restrict__ o,
                                     int K, int N) {
    int idx = blockIdx.x * 256 + threadIdx.x;     // over N*K, output [n][k]
    if (idx >= K * N) return;
    int n = idx / K, k = idx - n * K;
    o[idx] = (_Float16)a[(size_t)k * N + n];
}

// xe[i,d] = x[i,0]*w[0,d] + x[i,1]*w[1,d] + b[d]  (writes f32 + f16)
__global__ void embed_spec_kernel(const float* __restrict__ x, const float* __restrict__ w,
                                  const float* __restrict__ b, float* __restrict__ o32,
                                  _Float16* __restrict__ o16, int D) {
    size_t i = blockIdx.x;
    float x0 = x[i * 2], x1 = x[i * 2 + 1];
    for (int d = threadIdx.x; d < D; d += 256) {
        float v = x0 * w[d] + x1 * w[D + d] + b[d];
        o32[i * D + d] = v;
        o16[i * D + d] = (_Float16)v;
    }
}

// de[b,t,:] = emb[targets[b,t]]  (targets row stride Ty+1, use t in [0,Ty))
__global__ void gather_emb_kernel(const int* __restrict__ tgt, const float* __restrict__ emb,
                                  float* __restrict__ o32, _Float16* __restrict__ o16,
                                  int Ty, int D) {
    size_t i = blockIdx.x;                 // token index in [0, B*Ty)
    int b = (int)(i / Ty), t = (int)(i - (size_t)b * Ty);
    int id = tgt[(size_t)b * (Ty + 1) + t];
    const float* e = emb + (size_t)id * D;
    for (int d = threadIdx.x; d < D; d += 256) {
        float v = e[d];
        o32[i * D + d] = v;
        o16[i * D + d] = (_Float16)v;
    }
}

// fused residual add + LayerNorm over D=512, writes f32 + f16
__global__ __launch_bounds__(256)
void add_ln_kernel(const float* __restrict__ X, const float* __restrict__ Y,
                   const float* __restrict__ g, const float* __restrict__ bb,
                   float* __restrict__ O32, _Float16* __restrict__ O16, int D) {
    __shared__ float red[16];
    size_t row = blockIdx.x;
    const float* xr = X + row * D;
    const float* yr = Y + row * D;
    int tid = threadIdx.x;
    float v0 = xr[tid] + yr[tid];
    float v1 = xr[tid + 256] + yr[tid + 256];
    float s = v0 + v1, s2 = v0 * v0 + v1 * v1;
#pragma unroll
    for (int off = 16; off; off >>= 1) {
        s  += __shfl_xor(s,  off, 32);
        s2 += __shfl_xor(s2, off, 32);
    }
    int w = tid >> 5, lane = tid & 31;
    if (lane == 0) { red[w] = s; red[8 + w] = s2; }
    __syncthreads();
    if (tid == 0) {
        float ts = 0.f, ts2 = 0.f;
        for (int i = 0; i < 8; ++i) { ts += red[i]; ts2 += red[8 + i]; }
        red[0] = ts; red[1] = ts2;
    }
    __syncthreads();
    float mean = red[0] / (float)D;
    float var  = red[1] / (float)D - mean * mean;
    float inv  = rsqrtf(var + 1e-5f);
    float o0 = (v0 - mean) * inv * g[tid]       + bb[tid];
    float o1 = (v1 - mean) * inv * g[tid + 256] + bb[tid + 256];
    float* orow = O32 + row * D;
    orow[tid] = o0; orow[tid + 256] = o1;
    _Float16* orow16 = O16 + row * D;
    orow16[tid] = (_Float16)o0; orow16[tid + 256] = (_Float16)o1;
}

// ---------------------------------------------------------------------------
// WMMA GEMM: C[M,N](f32/f16) = A[M,K](f16,RM) * Bt[N,K](f16,RM) (+bias)(+relu)
// Double-buffered async global->LDS staging, 64-wide K-steps, 4 WMMA/wave/step.
// Requirements: K % 64 == 0, M % 64 == 0, Bt padded to a multiple of 64 rows.
// ---------------------------------------------------------------------------
template <bool BIASF, bool OUT32, bool OUT16, bool RELU, bool NGUARD>
__global__ __launch_bounds__(256)
void gemm_wmma_kernel(const _Float16* __restrict__ A, const _Float16* __restrict__ Bt,
                      const float* __restrict__ bias, float* __restrict__ C32,
                      _Float16* __restrict__ C16, int M, int N, int K) {
    __shared__ __attribute__((aligned(16))) _Float16 As[2][64][72];  // [buf][m][k]
    __shared__ __attribute__((aligned(16))) _Float16 Bs[2][64][72];  // [buf][n][k]
    const int tid  = threadIdx.x;
    const int lane = tid & 31;
    const int w    = tid >> 5;
    const int m0 = blockIdx.y * 64;
    const int n0 = blockIdx.x * 64;
    const int wm = (w & 3) * 16;      // wave m offset within tile
    const int wn = (w >> 2) * 32;     // wave n offset within tile (0 or 32)
    const int r  = lane & 15;
    const int kb = (lane >> 4) * 8;
    fvec8 acc0 = {}; fvec8 acc1 = {};

    const int row = tid >> 2;           // 0..63
    const int ch  = (tid & 3) * 16;     // 0,16,32,48
    const _Float16* ap = A  + (size_t)(m0 + row) * K + ch;
    const _Float16* bp = Bt + (size_t)(n0 + row) * K + ch;

    auto stage = [&](int buf, int k0) {
        copy16_g2l(ap + k0,     &As[buf][row][ch]);
        copy16_g2l(ap + k0 + 8, &As[buf][row][ch + 8]);
        copy16_g2l(bp + k0,     &Bs[buf][row][ch]);
        copy16_g2l(bp + k0 + 8, &Bs[buf][row][ch + 8]);
    };

    stage(0, 0);
    copy16_wait();
    __syncthreads();

    int buf = 0;
    for (int k0 = 0; k0 < K; k0 += 64) {
        if (k0 + 64 < K) stage(buf ^ 1, k0 + 64);   // prefetch next tile (other buffer)
#pragma unroll
        for (int kk = 0; kk < 64; kk += 32) {
            hvec16 a  = frag_ld(&As[buf][wm + r][kk + kb],
                                &As[buf][wm + r][kk + 16 + kb]);
            hvec16 b0 = frag_ld(&Bs[buf][wn + r][kk + kb],
                                &Bs[buf][wn + r][kk + 16 + kb]);
            hvec16 b1 = frag_ld(&Bs[buf][wn + 16 + r][kk + kb],
                                &Bs[buf][wn + 16 + r][kk + 16 + kb]);
            acc0 = wmma_f16(a, b0, acc0);
            acc1 = wmma_f16(a, b1, acc1);
        }
        copy16_wait();     // own async copies to the other buffer have landed
        __syncthreads();   // everyone done reading `buf` and done staging `buf^1`
        buf ^= 1;
    }

    // epilogue: lane holds n = lane&15, rows m = j + 8*(lane>>4); M % 64 == 0.
    const int half = lane >> 4;
    const int na = n0 + wn + r;
    const int nb = n0 + wn + 16 + r;
    float biasa = 0.f, biasb = 0.f;
    if (BIASF) {
        if (!NGUARD || na < N) biasa = bias[na];
        if (!NGUARD || nb < N) biasb = bias[nb];
    }
#pragma unroll
    for (int j = 0; j < 8; ++j) {
        const int m = m0 + wm + j + 8 * half;
        const size_t o = (size_t)m * N;
        float va = acc0[j] + biasa;
        float vb = acc1[j] + biasb;
        if (RELU) { va = fmaxf(va, 0.f); vb = fmaxf(vb, 0.f); }
        if (!NGUARD || na < N) {
            if (OUT32) C32[o + na] = va;
            if (OUT16) C16[o + na] = (_Float16)va;
        }
        if (!NGUARD || nb < N) {
            if (OUT32) C32[o + nb] = vb;
            if (OUT16) C16[o + nb] = (_Float16)vb;
        }
    }
    (void)M;
}

// ---------------------------------------------------------------------------
// Attention scores: S[bh, t, s] = Q[b,t,h,:] . K[b,s,h,:]   (scale folded into Wq)
// Scores buffer is (B*H, TP, SP) with TP = ceil(T/32)*32, so the t-store needs
// no guard; the s-store guard is a single wave-uniform branch.
// ---------------------------------------------------------------------------
__global__ __launch_bounds__(256)
void attn_scores_kernel(const _Float16* __restrict__ Q, const _Float16* __restrict__ Kk,
                        float* __restrict__ S, int T, int Sk, int TP, int SP, int H) {
    int bh = blockIdx.z;
    int b = bh / H, h = bh % H;
    int w = threadIdx.x >> 5, lane = threadIdx.x & 31;
    int t0 = blockIdx.y * 32 + (w >> 2) * 16;
    int s0 = blockIdx.x * 64 + (w & 3) * 16;
    int r  = lane & 15;
    int kb = (lane >> 4) * 8;
    int tq = t0 + r; if (tq > T - 1)  tq = T - 1;
    int sk = s0 + r; if (sk > Sk - 1) sk = Sk - 1;
    const _Float16* qrow = Q + ((size_t)(b * T + tq) * H + h) * 64;
    const _Float16* krow = Kk + ((size_t)(b * Sk + sk) * H + h) * 64;
    fvec8 acc = {};
#pragma unroll
    for (int k0 = 0; k0 < 64; k0 += 32) {
        hvec16 a  = frag_ld(qrow + k0 + kb, qrow + k0 + 16 + kb);
        hvec16 bf = frag_ld(krow + k0 + kb, krow + k0 + 16 + kb);
        acc = wmma_f16(a, bf, acc);
    }
    if (s0 < SP) {   // wave-uniform: 16-wide s-tile fully in or fully out (SP%32==0)
        float* srow = S + ((size_t)bh * TP + t0 + 8 * (lane >> 4)) * SP + s0 + r;
#pragma unroll
        for (int j = 0; j < 8; ++j) srow[(size_t)j * SP] = acc[j];
    }
}

// ---------------------------------------------------------------------------
// Row softmax (optionally causal). One wave per row; writes f16 probs with
// zero padding out to SP (multiple of 32) so the PV GEMM needs no K guard.
// ---------------------------------------------------------------------------
__global__ __launch_bounds__(256)
void softmax_kernel(const float* __restrict__ S, _Float16* __restrict__ P,
                    int T, int Sk, int TP, int SP, int causal) {
    int bh = blockIdx.y;
    int w = threadIdx.x >> 5, lane = threadIdx.x & 31;
    int t = blockIdx.x * 8 + w;
    if (t >= T) return;
    const float* srow = S + ((size_t)bh * TP + t) * SP;
    _Float16* prow = P + ((size_t)bh * TP + t) * SP;
    int nvalid = causal ? (t + 1) : Sk;
    float x[6];
    float mx = -3.4e38f;
#pragma unroll
    for (int i = 0; i < 6; ++i) {
        int s = lane + i * 32;
        float v = -3.4e38f;
        if (s < nvalid) v = srow[s];
        x[i] = v;
        mx = fmaxf(mx, v);
    }
#pragma unroll
    for (int off = 16; off; off >>= 1) mx = fmaxf(mx, __shfl_xor(mx, off, 32));
    float sum = 0.f;
#pragma unroll
    for (int i = 0; i < 6; ++i) {
        float e = (x[i] > -1e37f) ? __expf(x[i] - mx) : 0.f;
        x[i] = e;
        sum += e;
    }
#pragma unroll
    for (int off = 16; off; off >>= 1) sum += __shfl_xor(sum, off, 32);
    float inv = 1.f / sum;
#pragma unroll
    for (int i = 0; i < 6; ++i) {
        int s = lane + i * 32;
        if (s < SP) prow[s] = (_Float16)(x[i] * inv);
    }
}

// ---------------------------------------------------------------------------
// ctx[b,t,h,:] = sum_s P[bh,t,s] * V[b,s,h,:]
// V staged transposed in LDS so B-fragments are aligned ds_read_b128.
// ---------------------------------------------------------------------------
__global__ __launch_bounds__(256)
void attn_out_kernel(const _Float16* __restrict__ P, const _Float16* __restrict__ Vv,
                     _Float16* __restrict__ ctx, int T, int Sk, int TP, int SP, int H) {
    __shared__ __attribute__((aligned(16))) _Float16 Vt[64][200];  // [v][s], SP<=192
    int bh = blockIdx.y;
    int b = bh / H, h = bh % H;
    int tid = threadIdx.x, lane = tid & 31, w = tid >> 5;

    for (int idx = tid; idx < SP * 8; idx += 256) {
        int s = idx >> 3, vc = (idx & 7) * 8;
        hvec8 v = {};
        if (s < Sk) v = *(const hvec8*)(Vv + ((size_t)(b * Sk + s) * H + h) * 64 + vc);
#pragma unroll
        for (int j = 0; j < 8; ++j) Vt[vc + j][s] = v[j];
    }
    __syncthreads();

    int t0 = blockIdx.x * 128 + w * 16;
    int r  = lane & 15;
    int kb = (lane >> 4) * 8;
    int tq = t0 + r; if (tq > T - 1) tq = T - 1;
    const _Float16* prow = P + ((size_t)bh * TP + tq) * SP;

    fvec8 acc[4];
#pragma unroll
    for (int i = 0; i < 4; ++i) acc[i] = (fvec8){};

    for (int s0 = 0; s0 < SP; s0 += 32) {
        hvec16 a = frag_ld(prow + s0 + kb, prow + s0 + 16 + kb);
#pragma unroll
        for (int nt = 0; nt < 4; ++nt) {
            hvec16 bf = frag_ld(&Vt[nt * 16 + r][s0 + kb], &Vt[nt * 16 + r][s0 + 16 + kb]);
            acc[nt] = wmma_f16(a, bf, acc[nt]);
        }
    }
    int half = lane >> 4;
#pragma unroll
    for (int nt = 0; nt < 4; ++nt) {
#pragma unroll
        for (int j = 0; j < 8; ++j) {
            int t = t0 + j + 8 * half;
            if (t < T)
                ctx[((size_t)(b * T + t) * H + h) * 64 + nt * 16 + r] = (_Float16)acc[nt][j];
        }
    }
}

// ---------------------------------------------------------------------------
// Host-side orchestration
// ---------------------------------------------------------------------------
namespace {

struct Scratch {
    _Float16 *wq16, *wk16, *wv16, *wp16, *wff1, *wff2, *wll;
    float *actA32, *actB32, *sc32;
    _Float16 *actA16, *q16, *k16, *v16, *ctx16, *ff16, *enc16, *p16;
};

constexpr int Bb = 256, SXc = 180, TYc = 160, Dm = 512, Hh = 8, DFFc = 2048, Vc = 30;
constexpr int SPE = 192, SPD = 160;   // padded S widths (multiples of 32)
constexpr int TPE = 192, TPD = 160;   // padded T heights (multiples of 32)

inline void gemm_c16(const _Float16* A, const _Float16* Bt, _Float16* C16,
                     size_t M, int N, int K, hipStream_t stream) {
    dim3 grid((N + 63) / 64, (unsigned)((M + 63) / 64), 1);
    gemm_wmma_kernel<false, false, true, false, false>
        <<<grid, 256, 0, stream>>>(A, Bt, nullptr, nullptr, C16, (int)M, N, K);
}
inline void gemm_bias_c32(const _Float16* A, const _Float16* Bt, const float* bias,
                          float* C32, size_t M, int N, int K, hipStream_t stream) {
    dim3 grid((N + 63) / 64, (unsigned)((M + 63) / 64), 1);
    gemm_wmma_kernel<true, true, false, false, false>
        <<<grid, 256, 0, stream>>>(A, Bt, bias, C32, nullptr, (int)M, N, K);
}
inline void gemm_bias_relu_c16(const _Float16* A, const _Float16* Bt, const float* bias,
                               _Float16* C16, size_t M, int N, int K, hipStream_t stream) {
    dim3 grid((N + 63) / 64, (unsigned)((M + 63) / 64), 1);
    gemm_wmma_kernel<true, false, true, true, false>
        <<<grid, 256, 0, stream>>>(A, Bt, bias, nullptr, C16, (int)M, N, K);
}
inline void gemm_bias_c32_nguard(const _Float16* A, const _Float16* Bt, const float* bias,
                                 float* C32, size_t M, int N, int K, hipStream_t stream) {
    dim3 grid((N + 63) / 64, (unsigned)((M + 63) / 64), 1);
    gemm_wmma_kernel<true, true, false, false, true>
        <<<grid, 256, 0, stream>>>(A, Bt, bias, C32, nullptr, (int)M, N, K);
}

void run_attn_block(const float* wq, const float* wk, const float* wv,
                    const float* pw, const float* pb, const float* g, const float* bvec,
                    const _Float16* qin16, int Tq, int TP,
                    const _Float16* kin16, int Ts, int SP,
                    int causal,
                    const float* res32, float* out32, _Float16* out16,
                    const Scratch& ws, hipStream_t stream) {
    const size_t NQ = (size_t)Bb * Tq, NK = (size_t)Bb * Ts;
    const int nW = Dm * Dm;
    const float scale = 0.04419417382415922f;  // 1/sqrt(512): ref scales by sqrt(D)
    repack_qkv_kernel<<<(nW + 255) / 256, 256, 0, stream>>>(wq, ws.wq16, Hh, Dm, 64, scale);
    repack_qkv_kernel<<<(nW + 255) / 256, 256, 0, stream>>>(wk, ws.wk16, Hh, Dm, 64, 1.f);
    repack_qkv_kernel<<<(nW + 255) / 256, 256, 0, stream>>>(wv, ws.wv16, Hh, Dm, 64, 1.f);
    cvt_transpose_kernel<<<(nW + 255) / 256, 256, 0, stream>>>(pw, ws.wp16, Dm, Dm);

    gemm_c16(qin16, ws.wq16, ws.q16, NQ, Dm, Dm, stream);
    gemm_c16(kin16, ws.wk16, ws.k16, NK, Dm, Dm, stream);
    gemm_c16(kin16, ws.wv16, ws.v16, NK, Dm, Dm, stream);

    dim3 gs((Ts + 63) / 64, (Tq + 31) / 32, Bb * Hh);
    attn_scores_kernel<<<gs, 256, 0, stream>>>(ws.q16, ws.k16, ws.sc32,
                                               Tq, Ts, TP, SP, Hh);

    dim3 gsm((Tq + 7) / 8, Bb * Hh);
    softmax_kernel<<<gsm, 256, 0, stream>>>(ws.sc32, ws.p16, Tq, Ts, TP, SP, causal);

    dim3 go((Tq + 127) / 128, Bb * Hh);
    attn_out_kernel<<<go, 256, 0, stream>>>(ws.p16, ws.v16, ws.ctx16,
                                            Tq, Ts, TP, SP, Hh);

    gemm_bias_c32(ws.ctx16, ws.wp16, pb, ws.actB32, NQ, Dm, Dm, stream);
    add_ln_kernel<<<(unsigned)NQ, 256, 0, stream>>>(res32, ws.actB32, g, bvec,
                                                    out32, out16, Dm);
}

void run_ff_block(const float* w1, const float* b1, const float* w2, const float* b2,
                  const float* g, const float* bvec,
                  const _Float16* in16, const float* res32, size_t Ntok,
                  float* out32, _Float16* out16, const Scratch& ws, hipStream_t stream) {
    const size_t n1 = (size_t)Dm * DFFc;
    cvt_transpose_kernel<<<(unsigned)((n1 + 255) / 256), 256, 0, stream>>>(w1, ws.wff1,
                                                                           Dm, DFFc);
    cvt_transpose_kernel<<<(unsigned)((n1 + 255) / 256), 256, 0, stream>>>(w2, ws.wff2,
                                                                           DFFc, Dm);
    gemm_bias_relu_c16(in16, ws.wff1, b1, ws.ff16, Ntok, DFFc, Dm, stream);
    gemm_bias_c32(ws.ff16, ws.wff2, b2, ws.actB32, Ntok, Dm, DFFc, stream);
    add_ln_kernel<<<(unsigned)Ntok, 256, 0, stream>>>(res32, ws.actB32, g, bvec,
                                                      out32, out16, Dm);
}

}  // namespace

extern "C" void kernel_launch(void* const* d_in, const int* in_sizes, int n_in,
                              void* d_out, int out_size, void* d_ws, size_t ws_size,
                              hipStream_t stream) {
    (void)in_sizes; (void)n_in; (void)out_size; (void)ws_size;

    const float* x        = (const float*)d_in[0];
    const int*   targets  = (const int*)d_in[1];
    const float* spec_w   = (const float*)d_in[2];
    const float* spec_b   = (const float*)d_in[3];
    const float* pep_emb  = (const float*)d_in[4];
    const float* const* F = (const float* const*)d_in;
    const int ENC_AT = 5, DEC_SAT = 12, DEC_CAT = 19, ENC_FF = 26, DEC_FF = 32;
    const float* ll_w = (const float*)d_in[38];
    const float* ll_b = (const float*)d_in[39];
    float* out = (float*)d_out;

    const size_t NE = (size_t)Bb * SXc;   // 46080 encoder tokens
    const size_t ND = (size_t)Bb * TYc;   // 40960 decoder tokens
    const size_t TSPmax = (size_t)TPE * SPE;  // 192*192 padded score tile

    // --- carve workspace ---
    char* p = (char*)d_ws;
    auto alloc = [&](size_t bytes) -> char* {
        char* r = p;
        p += (bytes + 255) & ~(size_t)255;
        return r;
    };
    Scratch ws;
    ws.wq16  = (_Float16*)alloc((size_t)Dm * Dm * 2);
    ws.wk16  = (_Float16*)alloc((size_t)Dm * Dm * 2);
    ws.wv16  = (_Float16*)alloc((size_t)Dm * Dm * 2);
    ws.wp16  = (_Float16*)alloc((size_t)Dm * Dm * 2);
    ws.wff1  = (_Float16*)alloc((size_t)Dm * DFFc * 2);
    ws.wff2  = (_Float16*)alloc((size_t)Dm * DFFc * 2);
    ws.wll   = (_Float16*)alloc((size_t)64 * Dm * 2);   // padded to 64 rows (N=30)
    ws.actA32 = (float*)alloc(NE * Dm * 4);
    ws.actB32 = (float*)alloc(NE * Dm * 4);
    ws.actA16 = (_Float16*)alloc(NE * Dm * 2);
    ws.q16    = (_Float16*)alloc(NE * Dm * 2);
    ws.k16    = (_Float16*)alloc(NE * Dm * 2);
    ws.v16    = (_Float16*)alloc(NE * Dm * 2);
    ws.ctx16  = (_Float16*)alloc(NE * Dm * 2);
    ws.ff16   = (_Float16*)alloc(NE * DFFc * 2);
    ws.enc16  = (_Float16*)alloc(NE * Dm * 2);
    ws.sc32   = (float*)alloc((size_t)Bb * Hh * TSPmax * 4);
    ws.p16    = (_Float16*)alloc((size_t)Bb * Hh * TSPmax * 2);

    // ---------------- Encoder ----------------
    embed_spec_kernel<<<(unsigned)NE, 256, 0, stream>>>(x, spec_w, spec_b,
                                                        ws.actA32, ws.actA16, Dm);
    run_attn_block(F[ENC_AT], F[ENC_AT + 1], F[ENC_AT + 2], F[ENC_AT + 3], F[ENC_AT + 4],
                   F[ENC_AT + 5], F[ENC_AT + 6],
                   ws.actA16, SXc, TPE, ws.actA16, SXc, SPE, /*causal=*/0,
                   ws.actA32, ws.actA32, ws.actA16, ws, stream);
    run_ff_block(F[ENC_FF], F[ENC_FF + 1], F[ENC_FF + 2], F[ENC_FF + 3],
                 F[ENC_FF + 4], F[ENC_FF + 5],
                 ws.actA16, ws.actA32, NE, ws.actA32, ws.enc16, ws, stream);

    // ---------------- Decoder ----------------
    gather_emb_kernel<<<(unsigned)ND, 256, 0, stream>>>(targets, pep_emb,
                                                        ws.actA32, ws.actA16, TYc, Dm);
    run_attn_block(F[DEC_SAT], F[DEC_SAT + 1], F[DEC_SAT + 2], F[DEC_SAT + 3], F[DEC_SAT + 4],
                   F[DEC_SAT + 5], F[DEC_SAT + 6],
                   ws.actA16, TYc, TPD, ws.actA16, TYc, SPD, /*causal=*/1,
                   ws.actA32, ws.actA32, ws.actA16, ws, stream);
    run_attn_block(F[DEC_CAT], F[DEC_CAT + 1], F[DEC_CAT + 2], F[DEC_CAT + 3], F[DEC_CAT + 4],
                   F[DEC_CAT + 5], F[DEC_CAT + 6],
                   ws.actA16, TYc, TPD, ws.enc16, SXc, SPE, /*causal=*/0,
                   ws.actA32, ws.actA32, ws.actA16, ws, stream);
    run_ff_block(F[DEC_FF], F[DEC_FF + 1], F[DEC_FF + 2], F[DEC_FF + 3],
                 F[DEC_FF + 4], F[DEC_FF + 5],
                 ws.actA16, ws.actA32, ND, ws.actA32, ws.actA16, ws, stream);

    // ---------------- Logits ----------------
    const size_t nll = (size_t)Dm * Vc;
    cvt_transpose_kernel<<<(unsigned)((nll + 255) / 256), 256, 0, stream>>>(ll_w, ws.wll,
                                                                            Dm, Vc);
    gemm_bias_c32_nguard(ws.actA16, ws.wll, ll_b, out, ND, Vc, Dm, stream);
}
